// ZeroShotPredictor_42984032699048
// MI455X (gfx1250) — matrix operand order repaired and checked by reading, not compiled, not run on hardware
//
#include <hip/hip_runtime.h>
#include <hip/hip_bf16.h>

typedef __attribute__((ext_vector_type(16))) _Float16 v16h;
typedef __attribute__((ext_vector_type(8)))  float    v8f;
typedef __attribute__((ext_vector_type(4))) unsigned int u32x4;
typedef __attribute__((ext_vector_type(8))) int          i32x8;
typedef __attribute__((ext_vector_type(4))) int          i32x4;
typedef unsigned int u32;

#define N_ROWS   65536
#define K_KNOWN  300
#define M_NOVEL  200
#define D_EMB    300
#define D_PAD    320   // K padded to multiple of 32 for wmma f16 16x16x32
#define M_PAD    208   // N padded to multiple of 16
#define TOPK_T   10
#define PRE_T    0.5f
#define POST_T   0.1f
#define DETS     100

#define B_BYTES  (M_PAD * D_PAD * 2)     // 133120 bytes of B staged in LDS

// ---------------------------------------------------------------- kernel 0
// Normalize novel class embeddings -> f16 row-major [M_PAD][D_PAD], zero pad.
__global__ void k_novel_norm(const float* __restrict__ nce, _Float16* __restrict__ pB) {
  int wv = threadIdx.x >> 5, lane = threadIdx.x & 31;
  int m = blockIdx.x * 8 + wv;
  if (m >= M_PAD) return;
  float sv[10]; float ss = 0.f;
  if (m < M_NOVEL) {
#pragma unroll
    for (int j = 0; j < 10; ++j) {
      int d = lane + 32 * j;
      float v = (d < D_EMB) ? nce[(size_t)m * D_EMB + d] : 0.f;
      sv[j] = v; ss += v * v;
    }
#pragma unroll
    for (int off = 16; off; off >>= 1) ss += __shfl_xor(ss, off, 32);
    float rn = rsqrtf(ss);
#pragma unroll
    for (int j = 0; j < 10; ++j) {
      int d = lane + 32 * j;
      pB[(size_t)m * D_PAD + d] = (_Float16)(sv[j] * rn);
    }
  } else {
#pragma unroll
    for (int j = 0; j < 10; ++j) pB[(size_t)m * D_PAD + lane + 32 * j] = (_Float16)0.f;
  }
}

// ---------------------------------------------------------------- kernel 1
// One wave per proposal row: max, enable, sigmoid, top-10, weighted embedding,
// L2 normalize, write f16 A row [N][D_PAD] (zeroed when !enable).
__global__ void k_rows(const float* __restrict__ scores,
                       const float* __restrict__ kce,
                       const float* __restrict__ ilog,
                       _Float16* __restrict__ pA,
                       float* __restrict__ isig) {
  int wv = threadIdx.x >> 5, lane = threadIdx.x & 31;
  int r = blockIdx.x * 8 + wv;
  const float* srow = scores + (size_t)r * (K_KNOWN + 1);
  float sv[10];
  float mx = -1e30f;
#pragma unroll
  for (int j = 0; j < 10; ++j) {
    int d = lane + 32 * j;
    float v = (d < K_KNOWN) ? srow[d] : -1e30f;
    sv[j] = v; mx = fmaxf(mx, v);
  }
#pragma unroll
  for (int off = 16; off; off >>= 1) mx = fmaxf(mx, __shfl_xor(mx, off, 32));
  bool enable = (mx < PRE_T) && (mx > 0.001f);
  if (lane == 0) isig[r] = 1.f / (1.f + __expf(-ilog[r]));

  // iterative wave-wide argmax, ties -> lowest index (matches lax.top_k)
  float tv[TOPK_T]; int ti[TOPK_T];
  for (int t = 0; t < TOPK_T; ++t) {
    float bv = -1e30f; int bi = 0x7fffffff;
#pragma unroll
    for (int j = 0; j < 10; ++j) {
      int d = lane + 32 * j;
      if (sv[j] > bv || (sv[j] == bv && d < bi)) { bv = sv[j]; bi = d; }
    }
#pragma unroll
    for (int off = 16; off; off >>= 1) {
      float ov = __shfl_xor(bv, off, 32);
      int   oi = __shfl_xor(bi, off, 32);
      if (ov > bv || (ov == bv && oi < bi)) { bv = ov; bi = oi; }
    }
    tv[t] = bv; ti[t] = bi;
    if ((bi & 31) == lane) sv[bi >> 5] = -1e30f;   // knock out winner
  }
  float wsum = 0.f;
#pragma unroll
  for (int t = 0; t < TOPK_T; ++t) wsum += tv[t];
  float w[TOPK_T];
#pragma unroll
  for (int t = 0; t < TOPK_T; ++t) w[t] = tv[t] / wsum;

  float acc[10]; float ss = 0.f;
#pragma unroll
  for (int j = 0; j < 10; ++j) {
    int d = lane + 32 * j;
    float a = 0.f;
    if (d < D_EMB) {
#pragma unroll
      for (int t = 0; t < TOPK_T; ++t) a += w[t] * kce[(size_t)ti[t] * D_EMB + d];
    }
    acc[j] = a; ss += a * a;
  }
#pragma unroll
  for (int off = 16; off; off >>= 1) ss += __shfl_xor(ss, off, 32);
  float rn = enable ? rsqrtf(ss) : 0.f;   // !enable -> zero row -> sims row = 0
  _Float16* arow = pA + (size_t)r * D_PAD;
#pragma unroll
  for (int j = 0; j < 10; ++j) arow[lane + 32 * j] = (_Float16)(acc[j] * rn);
}

// ---------------------------------------------------------------- kernel 2
// WMMA GEMM: sims = A(65536xD_PAD f16) x B^T, B staged whole in LDS via the
// Tensor Data Mover, fused epilogue -> novel[N][M_PAD] f32.
__global__ void k_gemm(const _Float16* __restrict__ pA,
                       const _Float16* __restrict__ pB,
                       const float* __restrict__ isig,
                       const int* __restrict__ isper,
                       float* __restrict__ novel) {
  extern __shared__ _Float16 lB[];               // M_PAD*D_PAD halves = 133120 B
  int wv = threadIdx.x >> 5, lane = threadIdx.x & 31;
  int rt = blockIdx.x * 8 + wv;          // row tile [0,4096)
  int m  = lane & 15;
  int hi = lane >> 4;                    // lane half

  // ---- TDM: wave 0 DMA's the full B matrix (2D: 160 dwords x 208 rows) to LDS
  if (threadIdx.x < 32) {
    unsigned long long ga = (unsigned long long)(uintptr_t)pB;
    u32 ldsoff = (u32)(uintptr_t)lB;
    u32x4 g0 = { 1u,                                    // count=1, user descriptor
                 ldsoff,                                // lds_addr
                 (u32)ga,                               // global_addr[31:0]
                 (u32)(ga >> 32) | (2u << 30) };        // global_addr[56:32] | type=2
    i32x8 g1 = { (int)(2u << 16),                       // data_size=2 (4 bytes), mask=0
                 (int)(160u << 16),                     // tensor_dim0[15:0]=160
                 (int)(208u << 16),                     // tensor_dim0 hi=0 | tensor_dim1 lo=208
                 (int)(160u << 16),                     // tensor_dim1 hi=0 | tile_dim0=160
                 (int)208,                              // tile_dim1=208, tile_dim2=0
                 (int)160,                              // tensor_dim0_stride[31:0]=160
                 0, 0 };                                // stride hi bits / dim1_stride = 0
    i32x4 gz4 = { 0, 0, 0, 0 };
    i32x8 gz8 = { 0, 0, 0, 0, 0, 0, 0, 0 };
    __builtin_amdgcn_tensor_load_to_lds(g0, g1, gz4, gz4, gz8, 0);
    __builtin_amdgcn_s_wait_tensorcnt(0);
  }
  __syncthreads();

  // ---- A fragments: per ISA layout, each fragment is two contiguous 16B chunks
  union UA { v16h h; uint4 q[2]; };
  UA A[10];
  const uint4* aq = (const uint4*)(pA + (size_t)(rt * 16 + m) * D_PAD + (hi << 3));
#pragma unroll
  for (int f = 0; f < 10; ++f) {
    A[f].q[0] = aq[f * 4 + 0];           // halves 32f + 8*hi + [0..7]
    A[f].q[1] = aq[f * 4 + 2];           // halves 32f + 16 + 8*hi + [0..7]
  }
  float inter[8]; int iper[8];
#pragma unroll
  for (int rr = 0; rr < 8; ++rr) {
    int row = rt * 16 + rr + hi * 8;     // C/D layout: M = r + 8*lanehalf
    inter[rr] = isig[row];
    iper[rr]  = isper[row];
  }

  for (int ct = 0; ct < 13; ++ct) {
    int n = ct * 16 + m;                 // output column = novel class
    v8f c = {};
    const uint4* bq = (const uint4*)(lB + (size_t)n * D_PAD + (hi << 3));
#pragma unroll
    for (int f = 0; f < 10; ++f) {
      union UB { v16h h; uint4 q[2]; } B;
      B.q[0] = bq[f * 4 + 0];            // K = 32f + 8*hi + [0..7]
      B.q[1] = bq[f * 4 + 2];            // K = 32f + 16 + 8*hi + [0..7]
      c = __builtin_amdgcn_wmma_f32_16x16x32_f16(false, A[f].h, false, B.h,
                                                 (short)0, c, false, false);
    }
#pragma unroll
    for (int rr = 0; rr < 8; ++rr) {
      float v = c[rr] * inter[rr];
      v = (v < POST_T) ? 0.f : v;
      if (iper[rr] == 1) v = 0.f;
      novel[(size_t)(rt * 16 + rr + hi * 8) * M_PAD + ct * 16 + m] = v;
    }
  }
}

// ---------------------------------------------------------------- top-100 radix select
__global__ void k_init(u32* hist) { hist[threadIdx.x] = 0u; }

__global__ void k_hist(const float* __restrict__ vals, u32* __restrict__ hist,
                       const u32* __restrict__ pk, int shift) {
  __shared__ u32 lh[256];
  lh[threadIdx.x] = 0u;
  __syncthreads();
  u32 prefix = 0u, pmask = 0u;
  if (shift < 24) { prefix = pk[0]; pmask = 0xFFFFFFFFu << (shift + 8); }
  size_t n = (size_t)N_ROWS * M_PAD;
  size_t stride = (size_t)gridDim.x * blockDim.x;
  for (size_t i = (size_t)blockIdx.x * blockDim.x + threadIdx.x; i < n; i += stride) {
    u32 u = __float_as_uint(vals[i]);    // all values >= +0 -> uint order == float order
    if ((u & pmask) == prefix) atomicAdd(&lh[(u >> shift) & 255u], 1u);
  }
  __syncthreads();
  atomicAdd(&hist[threadIdx.x], lh[threadIdx.x]);
}

__global__ void k_select(u32* hist, u32* pk, int shift, int first) {
  if (threadIdx.x == 0) {
    u32 k      = first ? (u32)DETS : pk[1];
    u32 prefix = first ? 0u        : pk[0];
    int b;
    for (b = 255; b > 0; --b) {
      u32 cnt = hist[b];
      if (k <= cnt) break;
      k -= cnt;
    }
    prefix |= ((u32)b) << shift;
    pk[0] = prefix; pk[1] = k;
  }
  __syncthreads();
  hist[threadIdx.x] = 0u;   // ready for next pass
}

// ---------------------------------------------------------------- final assembly
__global__ void k_final(const float* __restrict__ scores,
                        const float* __restrict__ novel,
                        const u32* __restrict__ pk,
                        float* __restrict__ out) {
  int n = blockIdx.x;
  float thresh = __uint_as_float(pk[0]);  // exact 100th-largest value
  const float* srow = scores + (size_t)n * 301;
  float* orow = out + (size_t)n * 501;
  for (int cidx = threadIdx.x; cidx < 501; cidx += blockDim.x) {
    float v;
    if (cidx < 300)      v = srow[cidx];
    else if (cidx < 500) {
      float x = novel[(size_t)n * M_PAD + (cidx - 300)];
      v = (x <= thresh) ? 0.f : fminf(x * 3.f, 1.f);
    } else               v = srow[300];
    orow[cidx] = v;
  }
}

// ---------------------------------------------------------------- launch
extern "C" void kernel_launch(void* const* d_in, const int* in_sizes, int n_in,
                              void* d_out, int out_size, void* d_ws, size_t ws_size,
                              hipStream_t stream) {
  (void)in_sizes; (void)n_in; (void)out_size; (void)ws_size;
  const float* scores = (const float*)d_in[0];
  const float* deltas = (const float*)d_in[1];
  const float* kce    = (const float*)d_in[2];
  const float* nce    = (const float*)d_in[3];
  const float* ilog   = (const float*)d_in[4];
  const int*   isper  = (const int*)d_in[5];
  float* out = (float*)d_out;

  char* ws = (char*)d_ws;
  _Float16* pA   = (_Float16*)(ws);                       // 65536*320*2 = 41943040
  _Float16* pB   = (_Float16*)(ws + 41943040);            // 208*320*2   =   133120
  float*    isig = (float*)   (ws + 42076160);            // 65536*4     =   262144
  float*    novl = (float*)   (ws + 42338304);            // 65536*208*4 = 54525952
  u32*      hist = (u32*)     (ws + 96864256);            // 256*4
  u32*      pk   = (u32*)     (ws + 96865280);            // 2*4

  hipLaunchKernelGGL(k_novel_norm, dim3(26),    dim3(256), 0, stream, nce, pB);
  hipLaunchKernelGGL(k_rows,       dim3(8192),  dim3(256), 0, stream, scores, kce, ilog, pA, isig);
  hipLaunchKernelGGL(k_gemm,       dim3(512),   dim3(256), B_BYTES, stream, pA, pB, isig, isper, novl);
  hipLaunchKernelGGL(k_init,       dim3(1),     dim3(256), 0, stream, hist);
  for (int p = 0; p < 4; ++p) {
    int shift = 24 - 8 * p;
    hipLaunchKernelGGL(k_hist,   dim3(2048), dim3(256), 0, stream, novl, hist, pk, shift);
    hipLaunchKernelGGL(k_select, dim3(1),    dim3(256), 0, stream, hist, pk, shift, p == 0 ? 1 : 0);
  }
  hipLaunchKernelGGL(k_final, dim3(N_ROWS), dim3(256), 0, stream, scores, novl, pk, out);
  (void)hipMemcpyAsync(out + (size_t)N_ROWS * 501, deltas,
                       (size_t)N_ROWS * 4 * sizeof(float),
                       hipMemcpyDeviceToDevice, stream);
}
